// ScoreNet_10642928960064
// MI455X (gfx1250) — matrix operand order repaired and checked by reading, not compiled, not run on hardware
//
#include <hip/hip_runtime.h>
#include <hip/hip_bf16.h>
#include <math.h>

#define BB 8
#define NN 512
#define CC 384
#define HR 192
#define PH 16
#define KDIM 208
#define KPAD 224
#define HEAD_SCALE 0.17677669529663687f

typedef __bf16 bf16_t;
typedef bf16_t v16bf __attribute__((ext_vector_type(16)));
typedef float  v8f   __attribute__((ext_vector_type(8)));

struct Frag32 { uint4 lo; uint4 hi; };

__device__ __forceinline__ unsigned short f2bf(float f) {
    unsigned int u = __builtin_bit_cast(unsigned int, f);
    u += 0x7FFFu + ((u >> 16) & 1u);
    return (unsigned short)(u >> 16);
}

// Kernel 1: fold vertices + row projections -> bf16 fused-GEMM operands.
__global__ void fold_project_kernel(const float* __restrict__ feats,
                                    const float* __restrict__ Wq,
                                    const float* __restrict__ Wk,
                                    const float* __restrict__ Wpi,
                                    const float* __restrict__ bpi,
                                    const float* __restrict__ Wpj,
                                    const float* __restrict__ bpj,
                                    const float* __restrict__ alpha_pair,
                                    const float* __restrict__ Wd,
                                    const float* __restrict__ bd,
                                    unsigned short* __restrict__ Abuf,
                                    unsigned short* __restrict__ Bbuf,
                                    float* __restrict__ dvec) {
    const int row = blockIdx.x;
    const int b = row >> 9;
    const int n = row & (NN - 1);
    __shared__ float xs[CC];

    const float* f0 = feats + ((size_t)b * (2 * NN + 1) + 1 + 2 * n) * CC;
    for (int c = threadIdx.x; c < CC; c += blockDim.x)
        xs[c] = 0.5f * (f0[c] + f0[CC + c]);
    __syncthreads();

    const float ap = fmaxf(alpha_pair[0], 0.0f);
    unsigned short* arow = Abuf + (size_t)row * KPAD;
    unsigned short* brow = Bbuf + (size_t)row * KPAD;

    for (int o = threadIdx.x; o < 2 * HR + 2 * PH + 1; o += blockDim.x) {
        float acc = 0.0f;
        if (o < HR) {
            for (int c = 0; c < CC; ++c) acc += xs[c] * Wq[c * HR + o];
            arow[o] = f2bf(acc * HEAD_SCALE);
        } else if (o < 2 * HR) {
            const int oo = o - HR;
            for (int c = 0; c < CC; ++c) acc += xs[c] * Wk[c * HR + oo];
            brow[oo] = f2bf(acc);
        } else if (o < 2 * HR + PH) {
            const int oo = o - 2 * HR;
            acc = bpi[oo];
            for (int c = 0; c < CC; ++c) acc += xs[c] * Wpi[c * PH + oo];
            arow[HR + oo] = f2bf(fmaxf(acc, 0.0f) * ap);
        } else if (o < 2 * HR + 2 * PH) {
            const int oo = o - 2 * HR - PH;
            acc = bpj[oo];
            for (int c = 0; c < CC; ++c) acc += xs[c] * Wpj[c * PH + oo];
            brow[HR + oo] = f2bf(fmaxf(acc, 0.0f));
        } else {
            acc = bd[0];
            for (int c = 0; c < CC; ++c) acc += xs[c] * Wd[c];
            dvec[row] = acc;
        }
    }
    if (threadIdx.x < KPAD - KDIM) {
        arow[KDIM + threadIdx.x] = 0;
        brow[KDIM + threadIdx.x] = 0;
    }
}

// Kernel 2: per-batch covariance + power iteration -> cos/sin(theta0).
__global__ void orient_kernel(const float* __restrict__ xy,
                              float* __restrict__ thbuf) {
    const int b = blockIdx.x;
    const int t = threadIdx.x;
    __shared__ float r0[256], r1[256], r2[256];

    const float* p = xy + (size_t)b * NN * 2;
    float su = 0.f, sw = 0.f;
    for (int i = t; i < NN; i += 256) { su += p[2 * i]; sw += p[2 * i + 1]; }
    r0[t] = su; r1[t] = sw;
    __syncthreads();
    for (int s = 128; s > 0; s >>= 1) {
        if (t < s) { r0[t] += r0[t + s]; r1[t] += r1[t + s]; }
        __syncthreads();
    }
    const float mu = r0[0] / (float)NN;
    const float mw = r1[0] / (float)NN;
    __syncthreads();

    float cxx = 0.f, cxy = 0.f, cyy = 0.f;
    for (int i = t; i < NN; i += 256) {
        const float du = p[2 * i] - mu;
        const float dw = p[2 * i + 1] - mw;
        cxx += du * du; cxy += du * dw; cyy += dw * dw;
    }
    r0[t] = cxx; r1[t] = cxy; r2[t] = cyy;
    __syncthreads();
    for (int s = 128; s > 0; s >>= 1) {
        if (t < s) { r0[t] += r0[t + s]; r1[t] += r1[t + s]; r2[t] += r2[t + s]; }
        __syncthreads();
    }
    if (t == 0) {
        const float inv = 1.0f / ((float)NN + 1e-8f);
        const float Cxx = r0[0] * inv, Cxy = r1[0] * inv, Cyy = r2[0] * inv;
        float v0 = 1.0f, v1 = 0.0f;
        #pragma unroll
        for (int it = 0; it < 5; ++it) {
            const float n0 = Cxx * v0 + Cxy * v1;
            const float n1 = Cxy * v0 + Cyy * v1;
            const float nn = sqrtf(n0 * n0 + n1 * n1) + 1e-8f;
            v0 = n0 / nn; v1 = n1 / nn;
        }
        thbuf[2 * b] = v0;
        thbuf[2 * b + 1] = v1;
    }
}

// Kernel 3: one wave = one 16x16 tile; v_wmma_f32_16x16x32_bf16 GEMM + fused MLPs.
__global__ void score_kernel(const unsigned short* __restrict__ Abuf,
                             const unsigned short* __restrict__ Bbuf,
                             const float* __restrict__ dvec,
                             const float* __restrict__ thbuf,
                             const float* __restrict__ xy,
                             const float* __restrict__ Wg1,
                             const float* __restrict__ bg1,
                             const float* __restrict__ Wg2,
                             const float* __restrict__ bg2,
                             const float* __restrict__ alpha_geom,
                             const float* __restrict__ Wa1,
                             const float* __restrict__ ba1,
                             const float* __restrict__ Wa2,
                             const float* __restrict__ ba2,
                             const float* __restrict__ alpha_angle,
                             const float* __restrict__ logit_scale,
                             float* __restrict__ out) {
    __shared__ float sWg1[12 * 32], sbg1[32], sWg2[32];
    __shared__ float sWa1[6 * 16], sba1[16], sWa2[16];
    __shared__ float sc[5];

    const int t = threadIdx.x;                  // 128 threads = 4 waves
    for (int i = t; i < 12 * 32; i += 128) sWg1[i] = Wg1[i];
    if (t < 32) { sbg1[t] = bg1[t]; sWg2[t] = Wg2[t]; }
    for (int i = t; i < 6 * 16; i += 128) sWa1[i] = Wa1[i];
    if (t < 16) { sba1[t] = ba1[t]; sWa2[t] = Wa2[t]; }
    if (t == 0) {
        sc[0] = bg2[0];
        sc[1] = ba2[0];
        sc[2] = fmaxf(alpha_geom[0], 0.0f);
        sc[3] = fmaxf(alpha_angle[0], 0.0f);
        sc[4] = fmaxf(logit_scale[0], 0.01f);
    }
    __syncthreads();

    const int wid  = t >> 5;
    const int lane = t & 31;
    const int tile = blockIdx.x * 4 + wid;
    const int b  = tile >> 10;
    const int rr = tile & 1023;
    const int tn = rr >> 5;
    const int tm = rr & 31;

    const int colN = lane & 15;
    const int half = lane >> 4;

    const unsigned short* Arow = Abuf + (size_t)(b * NN + tn * 16 + colN) * KPAD;
    const unsigned short* Brow = Bbuf + (size_t)(b * NN + tm * 16 + colN) * KPAD;
    const int akb = half * 8;
    const int bkb = half * 8;

    v8f acc = {};
    // unroll 2: keeps <=4 fragments live (no VGPR spill) while overlapping
    // fragment loads with WMMA issue.
    #pragma unroll 2
    for (int kc = 0; kc < KPAD / 32; ++kc) {
        const int k0 = kc * 32;
        Frag32 ar, br;
        ar.lo = *(const uint4*)(Arow + k0 + akb);
        ar.hi = *(const uint4*)(Arow + k0 + akb + 16);
        br.lo = *(const uint4*)(Brow + k0 + bkb);
        br.hi = *(const uint4*)(Brow + k0 + bkb + 16);
        const v16bf af = __builtin_bit_cast(v16bf, ar);
        const v16bf bf = __builtin_bit_cast(v16bf, br);
        acc = __builtin_amdgcn_wmma_f32_16x16x32_bf16(
            false, af, false, bf, (short)0, acc, false, false);
    }

    const int m = tm * 16 + colN;
    const float um = xy[((size_t)b * NN + m) * 2 + 0];
    const float wm = xy[((size_t)b * NN + m) * 2 + 1];
    const float c0 = thbuf[2 * b], s0 = thbuf[2 * b + 1];
    const float bg2v = sc[0], ba2v = sc[1];
    const float ageom = sc[2], aangle = sc[3], lscale = sc[4];

    const float sp    = 1.4142135623730951f / 7.0f;
    const float gamma = 1.0f / (2.0f * sp * sp + 1e-8f);

    for (int v = 0; v < 8; ++v) {
        const int nrow = tn * 16 + v + half * 8;
        const float un = xy[((size_t)b * NN + nrow) * 2 + 0];
        const float wn = xy[((size_t)b * NN + nrow) * 2 + 1];
        const float dy = un - um;
        const float dx = wn - wm;
        const float h2 = dy * dy + dx * dx;
        float cth, sth;
        if (h2 > 0.0f) {
            const float hinv = rsqrtf(h2);
            cth = dx * hinv; sth = dy * hinv;
        } else { cth = 1.0f; sth = 0.0f; }
        const float r = sqrtf(h2 + 1e-8f);
        const float c1  = cth * c0 + sth * s0;
        const float s1v = sth * c0 - cth * s0;

        float gf[12];
        gf[0] = dy; gf[1] = dx;
        #pragma unroll
        for (int k = 0; k < 8; ++k) {
            const float dmu = r - (float)k * sp;
            gf[2 + k] = __expf(-gamma * dmu * dmu);
        }
        gf[10] = c1; gf[11] = s1v;

        float G = bg2v;
        #pragma unroll 4
        for (int h = 0; h < 32; ++h) {
            float a = sbg1[h];
            #pragma unroll
            for (int j = 0; j < 12; ++j) a += gf[j] * sWg1[j * 32 + h];
            G += sWg2[h] * fmaxf(a, 0.0f);
        }

        const float c2 = c1 * c1 - s1v * s1v, s2 = 2.0f * c1 * s1v;
        const float c4 = c2 * c2 - s2 * s2,   s4 = 2.0f * c2 * s2;
        const float af6[6] = { c1, s1v, c2, s2, c4, s4 };
        float Hb = ba2v;
        #pragma unroll 4
        for (int h = 0; h < 16; ++h) {
            float a2 = sba1[h];
            #pragma unroll
            for (int j = 0; j < 6; ++j) a2 += af6[j] * sWa1[j * 16 + h];
            Hb += sWa2[h] * fmaxf(a2, 0.0f);
        }

        float s = acc[v] + ageom * G + aangle * Hb;
        if (nrow == m) s += dvec[b * NN + nrow];
        out[((size_t)b * NN + nrow) * NN + m] = s * lscale;
    }
}

extern "C" void kernel_launch(void* const* d_in, const int* in_sizes, int n_in,
                              void* d_out, int out_size, void* d_ws, size_t ws_size,
                              hipStream_t stream) {
    const float* feats       = (const float*)d_in[0];
    const float* xy          = (const float*)d_in[1];
    const float* Wq          = (const float*)d_in[2];
    const float* Wk          = (const float*)d_in[3];
    const float* Wpi         = (const float*)d_in[4];
    const float* bpi         = (const float*)d_in[5];
    const float* Wpj         = (const float*)d_in[6];
    const float* bpj         = (const float*)d_in[7];
    const float* alpha_pair  = (const float*)d_in[8];
    const float* Wg1         = (const float*)d_in[9];
    const float* bg1         = (const float*)d_in[10];
    const float* Wg2         = (const float*)d_in[11];
    const float* bg2         = (const float*)d_in[12];
    const float* alpha_geom  = (const float*)d_in[13];
    const float* Wa1         = (const float*)d_in[14];
    const float* ba1         = (const float*)d_in[15];
    const float* Wa2         = (const float*)d_in[16];
    const float* ba2         = (const float*)d_in[17];
    const float* alpha_angle = (const float*)d_in[18];
    const float* Wd          = (const float*)d_in[19];
    const float* bd          = (const float*)d_in[20];
    const float* logit_scale = (const float*)d_in[21];
    float* out = (float*)d_out;

    char* ws = (char*)d_ws;
    const size_t szAB = (size_t)BB * NN * KPAD * sizeof(unsigned short);
    unsigned short* Abuf = (unsigned short*)(ws);
    unsigned short* Bbuf = (unsigned short*)(ws + szAB);
    float* dvec  = (float*)(ws + 2 * szAB);
    float* thbuf = (float*)(ws + 2 * szAB + (size_t)BB * NN * sizeof(float));

    fold_project_kernel<<<BB * NN, 256, 0, stream>>>(
        feats, Wq, Wk, Wpi, bpi, Wpj, bpj, alpha_pair, Wd, bd,
        Abuf, Bbuf, dvec);

    orient_kernel<<<BB, 256, 0, stream>>>(xy, thbuf);

    score_kernel<<<(BB * (NN / 16) * (NN / 16)) / 4, 128, 0, stream>>>(
        Abuf, Bbuf, dvec, thbuf, xy,
        Wg1, bg1, Wg2, bg2, alpha_geom,
        Wa1, ba1, Wa2, ba2, alpha_angle,
        logit_scale, out);
}
